// Agent2Lane_emb_attn_30537217474894
// MI455X (gfx1250) — matrix operand order repaired and checked by reading, not compiled, not run on hardware
//
#include <hip/hip_runtime.h>
#include <hip/hip_bf16.h>
#include <stdint.h>

// ---------------------------------------------------------------------------
// Problem constants (from reference): B=2, N=64, M=64, L=30, E=128, FA=16, H=4
// ---------------------------------------------------------------------------
#define B_DIM 2
#define N_DIM 64
#define M_DIM 64
#define L_DIM 30
#define E_DIM 128
#define H_DIM 4
#define DH_DIM 32

typedef __attribute__((ext_vector_type(16))) __bf16    v16bf;
typedef __attribute__((ext_vector_type(8)))  float     v8f;
typedef __attribute__((ext_vector_type(8)))  uint32_t  v8u;

union FragB { v8u u; v16bf b; };

__device__ __forceinline__ uint16_t f32_to_bf16(float f) {
  uint32_t u = __builtin_bit_cast(uint32_t, f);
  u += 0x7FFFu + ((u >> 16) & 1u);      // round-to-nearest-even
  return (uint16_t)(u >> 16);
}

// ---------------------------------------------------------------------------
// Kernel W: weight preprocessing.
//   WoT[n][k] = bf16(Wo[k][n]),  WpT[n][k] = bf16(Wp[k][n])  (B-fragment layout:
//   K contiguous per output column n), WvT[n][k] = Wv[k][n] in f32.
// ---------------------------------------------------------------------------
__global__ void prep_weights(const float* __restrict__ Wv,
                             const float* __restrict__ Wo,
                             const float* __restrict__ Wp,
                             uint16_t* __restrict__ WoT,
                             uint16_t* __restrict__ WpT,
                             float* __restrict__ WvT) {
  int i = blockIdx.x * blockDim.x + threadIdx.x;   // 0..16383
  int n = i >> 7, k = i & 127;
  WoT[n * 128 + k] = f32_to_bf16(Wo[k * 128 + n]);
  WpT[n * 128 + k] = f32_to_bf16(Wp[k * 128 + n]);
  WvT[n * 128 + k] = Wv[k * 128 + n];
}

// ---------------------------------------------------------------------------
// Kernel Q: per-(b,n) query pipeline.
//   afeat -> ae -> q; then fold Wk through q:
//   kq[h][e] = (1/sqrt(dh)) * sum_d Wk[e][h*32+d] * q[h*32+d]
//   cq[h]    = (1/sqrt(dh)) * sum_d bk[h*32+d]    * q[h*32+d]
// ---------------------------------------------------------------------------
__global__ void prep_query(const float* __restrict__ agent,
                           const float* __restrict__ Wa, const float* __restrict__ ba,
                           const float* __restrict__ Wq, const float* __restrict__ bq,
                           const float* __restrict__ Wk, const float* __restrict__ bk,
                           float* __restrict__ KQ, float* __restrict__ CQ) {
  int bn = blockIdx.x;        // 0..B*N-1
  int e  = threadIdx.x;       // 0..127
  __shared__ float ae[E_DIM];
  __shared__ float q[E_DIM];

  const float* ag = agent + (size_t)bn * 16;
  float af[12];
  af[0] = ag[2];
  #pragma unroll
  for (int j = 0; j < 11; ++j) af[1 + j] = ag[5 + j];

  float acc = ba[e];
  #pragma unroll
  for (int j = 0; j < 12; ++j) acc += af[j] * Wa[j * 128 + e];
  ae[e] = acc;
  __syncthreads();

  acc = bq[e];
  for (int j = 0; j < 128; ++j) acc += ae[j] * Wq[j * 128 + e];
  q[e] = acc;
  __syncthreads();

  const float scale = 0.17677669529663688f;  // 1/sqrt(32)
  #pragma unroll
  for (int h = 0; h < H_DIM; ++h) {
    float s = 0.f;
    const float* wrow = Wk + (size_t)e * 128 + h * DH_DIM;
    for (int d = 0; d < DH_DIM; ++d) s += wrow[d] * q[h * DH_DIM + d];
    KQ[((size_t)bn * H_DIM + h) * 128 + e] = s * scale;
  }
  if (e < H_DIM) {
    float s = 0.f;
    for (int d = 0; d < DH_DIM; ++d) s += bk[e * DH_DIM + d] * q[e * DH_DIM + d];
    CQ[bn * H_DIM + e] = s * scale;
  }
}

// ---------------------------------------------------------------------------
// Kernel A: one block per (b,n,m). Edge rel-pose -> ee (LDS only) -> scores
// (via folded kq) -> softmax -> ctx -> attn = ctx @ Wv + bv, stored as bf16.
// ---------------------------------------------------------------------------
#define EESTR 132   // padded row stride (floats) to avoid 64-bank conflicts

__global__ void __launch_bounds__(128)
attn_kernel(const float* __restrict__ agent, const float* __restrict__ lane,
            const float* __restrict__ We, const float* __restrict__ be,
            const float* __restrict__ pe,
            const float* __restrict__ KQ, const float* __restrict__ CQ,
            const float* __restrict__ WvT, const float* __restrict__ bv,
            uint16_t* __restrict__ ATT) {
  int blk = blockIdx.x;         // 0..B*N*M-1
  int m   = blk & (M_DIM - 1);
  int bn  = blk >> 6;
  int b   = bn >> 6;
  int t   = threadIdx.x;        // 0..127

  __shared__ float edge[L_DIM][4];
  __shared__ float ee[L_DIM * EESTR];
  __shared__ float kq[H_DIM * E_DIM];
  __shared__ float cq[H_DIM];
  __shared__ float sw[H_DIM * L_DIM];
  __shared__ float ctx[H_DIM * E_DIM];

  for (int i = t; i < H_DIM * E_DIM; i += 128) kq[i] = KQ[(size_t)bn * H_DIM * E_DIM + i];
  if (t < H_DIM) cq[t] = CQ[bn * H_DIM + t];

  if (t < L_DIM) {
    const float* ag = agent + (size_t)bn * 16;
    float x = ag[0], y = ag[1], s = ag[3], c = ag[4];
    float f1 = (x == 0.f && y == 0.f && s == 0.f && c == 0.f) ? 0.f : 1.f;
    const float* p = lane + ((size_t)b * M_DIM + m) * (4 * L_DIM) + t * 4;
    float px = p[0], py = p[1], ps = p[2], pc = p[3];
    float f2 = (px == 0.f && py == 0.f && ps == 0.f && pc == 0.f) ? 0.f : 1.f;
    float g  = f1 * f2;
    float dx = px - x, dy = py - y;
    edge[t][0] = g * (c * dx + s * dy) * 0.1f;
    edge[t][1] = g * (c * dy - s * dx) * 0.1f;
    edge[t][2] = g * (ps * c - pc * s);
    edge[t][3] = g * (pc * c + ps * s);
  }
  __syncthreads();

  {  // ee[l][e], thread t owns column e = t
    int e = t;
    float w0 = We[e], w1 = We[128 + e], w2 = We[256 + e], w3 = We[384 + e];
    float b0 = be[e];
    for (int l = 0; l < L_DIM; ++l) {
      ee[l * EESTR + e] = b0 + pe[l * 128 + e] +
                          edge[l][0] * w0 + edge[l][1] * w1 +
                          edge[l][2] * w2 + edge[l][3] * w3;
    }
  }
  __syncthreads();

  if (t < H_DIM * L_DIM) {  // scores[h][l] = ee[l] . kq[h] + cq[h]
    int h = t / L_DIM, l = t % L_DIM;
    float s = cq[h];
    const float* kqh = &kq[h * E_DIM];
    const float* eel = &ee[l * EESTR];
    for (int e = 0; e < E_DIM; ++e) s += eel[e] * kqh[e];
    sw[t] = s;
  }
  __syncthreads();

  if (t < H_DIM) {  // softmax over L per head
    float mx = -1e30f;
    for (int l = 0; l < L_DIM; ++l) mx = fmaxf(mx, sw[t * L_DIM + l]);
    float sum = 0.f;
    for (int l = 0; l < L_DIM; ++l) {
      float v = __expf(sw[t * L_DIM + l] - mx);
      sw[t * L_DIM + l] = v;
      sum += v;
    }
    float inv = 1.f / sum;
    for (int l = 0; l < L_DIM; ++l) sw[t * L_DIM + l] *= inv;
  }
  __syncthreads();

  {  // ctx[h][e] = sum_l w[h][l] * ee[l][e]
    int e = t;
    #pragma unroll
    for (int h = 0; h < H_DIM; ++h) {
      float s = 0.f;
      for (int l = 0; l < L_DIM; ++l) s += sw[h * L_DIM + l] * ee[l * EESTR + e];
      ctx[h * E_DIM + e] = s;
    }
  }
  __syncthreads();

  {  // attn[e'] = ctx[head(e')] . Wv[:,e'] + bv[e']   (WvT rows are contiguous)
    int ep = t, h = ep >> 5;
    float s = bv[ep];
    const float* ch = &ctx[h * E_DIM];
    const float* wr = &WvT[(size_t)ep * 128];
    for (int e = 0; e < E_DIM; ++e) s += ch[e] * wr[e];
    ATT[(size_t)blk * E_DIM + ep] = f32_to_bf16(s);
  }
}

// ---------------------------------------------------------------------------
// Kernel P: fused double projection OUT = ((ATT @ Wo)+bo) @ Wp + bp
// One wave (32 lanes) per 16x128 row tile, v_wmma_f32_16x16x32_bf16.
// Fragment layouts used here (A/B: contiguous K order, same in both):
//   A : lane l -> M = l%16, K in hs*16 + 0..15 of each 32-K step
//   B : lane l -> N = l%16, K in hs*16 + 0..15 of each 32-K step
//   C/D: vgpr r, lane l -> M = r + 8*hs, N = l%16
// ---------------------------------------------------------------------------
#define TSTR 132   // padded LDS tile row stride (halfwords)

__global__ void __launch_bounds__(32)
proj_kernel(const uint16_t* __restrict__ ATT,
            const uint16_t* __restrict__ WoT, const float* __restrict__ bo,
            const uint16_t* __restrict__ WpT, const float* __restrict__ bp,
            float* __restrict__ OUT) {
  __shared__ uint16_t tile[16 * TSTR];

  int lane    = threadIdx.x & 31;
  int mrow    = lane & 15;
  int hs      = lane >> 4;
  int rowbase = blockIdx.x * 16;

  // ---- A1 fragments straight from global (ATT is L2-resident bf16) ----
  FragB a1[4];
  const uint16_t* arow = ATT + (size_t)(rowbase + mrow) * E_DIM;
  #pragma unroll
  for (int kk = 0; kk < 4; ++kk) {
    #pragma unroll
    for (int d = 0; d < 8; ++d) {
      int k0 = kk * 32 + hs * 16 + d * 2;
      a1[kk].u[d] = *(const uint32_t*)(arow + k0);
    }
  }

  // ---- GEMM 1: tmp = A1 @ Wo + bo, re-laid into LDS as bf16 A-tile ----
  for (int j = 0; j < 8; ++j) {
    v8f acc = {0.f, 0.f, 0.f, 0.f, 0.f, 0.f, 0.f, 0.f};
    #pragma unroll
    for (int kk = 0; kk < 4; ++kk) {
      FragB bf;
      const uint16_t* bcol = WoT + (size_t)(j * 16 + mrow) * 128 + kk * 32 + hs * 16;
      #pragma unroll
      for (int d = 0; d < 8; ++d) bf.u[d] = *(const uint32_t*)(bcol + d * 2);
      acc = __builtin_amdgcn_wmma_f32_16x16x32_bf16(
          false, a1[kk].b, false, bf.b, (short)0, acc, false, false);
    }
    float bias = bo[j * 16 + mrow];
    #pragma unroll
    for (int r = 0; r < 8; ++r) {
      tile[(r + 8 * hs) * TSTR + j * 16 + mrow] = f32_to_bf16(acc[r] + bias);
    }
  }
  __syncthreads();

  // ---- Gather A2 fragments from LDS (padded stride: no bank conflicts) ----
  FragB a2[4];
  #pragma unroll
  for (int kk = 0; kk < 4; ++kk) {
    #pragma unroll
    for (int d = 0; d < 8; ++d) {
      int k0 = kk * 32 + hs * 16 + d * 2;
      a2[kk].u[d] = *(const uint32_t*)&tile[mrow * TSTR + k0];
    }
  }

  // ---- GEMM 2: OUT = A2 @ Wp + bp ----
  for (int j = 0; j < 8; ++j) {
    v8f acc = {0.f, 0.f, 0.f, 0.f, 0.f, 0.f, 0.f, 0.f};
    #pragma unroll
    for (int kk = 0; kk < 4; ++kk) {
      FragB bf;
      const uint16_t* bcol = WpT + (size_t)(j * 16 + mrow) * 128 + kk * 32 + hs * 16;
      #pragma unroll
      for (int d = 0; d < 8; ++d) bf.u[d] = *(const uint32_t*)(bcol + d * 2);
      acc = __builtin_amdgcn_wmma_f32_16x16x32_bf16(
          false, a2[kk].b, false, bf.b, (short)0, acc, false, false);
    }
    float bias = bp[j * 16 + mrow];
    #pragma unroll
    for (int r = 0; r < 8; ++r) {
      OUT[(size_t)(rowbase + r + 8 * hs) * E_DIM + j * 16 + mrow] = acc[r] + bias;
    }
  }
}

// ---------------------------------------------------------------------------
// Launch
// ---------------------------------------------------------------------------
extern "C" void kernel_launch(void* const* d_in, const int* in_sizes, int n_in,
                              void* d_out, int out_size, void* d_ws, size_t ws_size,
                              hipStream_t stream) {
  const float* agent = (const float*)d_in[0];
  const float* lane  = (const float*)d_in[1];
  const float* We    = (const float*)d_in[2];
  const float* be    = (const float*)d_in[3];
  const float* Wa    = (const float*)d_in[4];
  const float* ba    = (const float*)d_in[5];
  const float* pe    = (const float*)d_in[6];
  const float* Wq    = (const float*)d_in[7];
  const float* bq    = (const float*)d_in[8];
  const float* Wk    = (const float*)d_in[9];
  const float* bk    = (const float*)d_in[10];
  const float* Wv    = (const float*)d_in[11];
  const float* bv    = (const float*)d_in[12];
  const float* Wo    = (const float*)d_in[13];
  const float* bo    = (const float*)d_in[14];
  const float* Wp    = (const float*)d_in[15];
  const float* bp    = (const float*)d_in[16];

  char* ws = (char*)d_ws;
  uint16_t* WoT = (uint16_t*)(ws + 0);            //  32 KB
  uint16_t* WpT = (uint16_t*)(ws + 32768);        //  32 KB
  float*    WvT = (float*)   (ws + 65536);        //  64 KB
  float*    KQ  = (float*)   (ws + 131072);       // 256 KB
  float*    CQ  = (float*)   (ws + 393216);       //   2 KB
  uint16_t* ATT = (uint16_t*)(ws + 395264);       //   2 MB

  prep_weights<<<64, 256, 0, stream>>>(Wv, Wo, Wp, WoT, WpT, WvT);
  prep_query<<<B_DIM * N_DIM, 128, 0, stream>>>(agent, Wa, ba, Wq, bq, Wk, bk, KQ, CQ);
  attn_kernel<<<B_DIM * N_DIM * M_DIM, 128, 0, stream>>>(agent, lane, We, be, pe,
                                                         KQ, CQ, WvT, bv, ATT);
  proj_kernel<<<(B_DIM * N_DIM * M_DIM) / 16, 32, 0, stream>>>(ATT, WoT, bo, WpT, bp,
                                                               (float*)d_out);
}
